// GATEncoder_72541997629771
// MI455X (gfx1250) — matrix operand order repaired and checked by reading, not compiled, not run on hardware
//
#include <hip/hip_runtime.h>

// ---------------------------------------------------------------------------
// GATEncoder for gfx1250 (MI455X).  Dense projections use
// v_wmma_f32_16x16x32_bf16 (fp32 accumulate); edge softmax/aggregation are
// bandwidth-bound scatter/gather kernels.
// ---------------------------------------------------------------------------

typedef unsigned short u16;
typedef __bf16 bf16;
typedef bf16  v16bf __attribute__((ext_vector_type(16)));
typedef float v8f   __attribute__((ext_vector_type(8)));

#define FDIM  256   // K and N_out of every GEMM (Fin = H*Hd = H*L*? = 256)
#define HEADS 4
#define NEG_SLOPE 0.2f
#define BN_EPS 1e-5f

union Frag { uint4 u[2]; v16bf v; };

// float -> bf16 bits, round-to-nearest-even
__device__ __forceinline__ u16 f2bf(float f) {
    unsigned u = __float_as_uint(f);
    u += 0x7FFFu + ((u >> 16) & 1u);
    return (u16)(u >> 16);
}
// order-preserving float<->uint transform (for atomicMax-based segment max)
__device__ __forceinline__ unsigned f2ord(float f) {
    unsigned b = __float_as_uint(f);
    return (b & 0x80000000u) ? ~b : (b | 0x80000000u);
}
__device__ __forceinline__ float ord2f(unsigned k) {
    unsigned b = (k & 0x80000000u) ? (k & 0x7FFFFFFFu) : ~k;
    return __uint_as_float(b);
}
__device__ __forceinline__ float dot8(float4 a0, float4 a1, float4 b0, float4 b1) {
    return a0.x*b0.x + a0.y*b0.y + a0.z*b0.z + a0.w*b0.w
         + a1.x*b1.x + a1.y*b1.y + a1.z*b1.z + a1.w*b1.w;
}

// --------------------------- precision conversion ---------------------------
__global__ void cvt_bf16(const float* __restrict__ in, u16* __restrict__ out, int total) {
    int tid = blockIdx.x * blockDim.x + threadIdx.x;
    if (tid < total) out[tid] = f2bf(in[tid]);
}

// Repack W[K=256][N=256] (row-major fp32) into the native wave32 B-fragment
// layout for V_WMMA_F32_16X16X32_BF16:
//   fragment (nt,kt): lane n(0..15) holds column nt*16+n, K = kt*32 + {0..7, 16..23};
//                     lane n+16     holds same column,   K = kt*32 + {8..15, 24..31}.
// Each lane's 16 bf16 are stored contiguously (32B) -> coalesced uint4 loads.
__global__ void pack_w(const float* __restrict__ W, u16* __restrict__ Wp) {
    int tid = blockIdx.x * blockDim.x + threadIdx.x;   // 16*8*32*16 = 65536
    int j    =  tid        & 15;
    int lane = (tid >> 4)  & 31;
    int kt   = (tid >> 9)  & 7;
    int nt   =  tid >> 12;
    int n = nt * 16 + (lane & 15);
    int k = kt * 32 + (lane >> 4) * 8 + (j & 7) + (j >> 3) * 16;
    Wp[tid] = f2bf(W[k * FDIM + n]);
}

// ------------------------------- WMMA GEMM ----------------------------------
// C[M,256] = A[M,256] @ W[256,256], A/B bf16, C fp32.
// Block = 256 threads = 8 waves; block handles one 16-row M tile; wave w
// computes N tiles 2w and 2w+1 (A fragment reused across both, W stays in L2).
__global__ void gemm_bf16_wmma(const u16* __restrict__ A,
                               const u16* __restrict__ Bp,
                               float* __restrict__ C) {
    int wave = threadIdx.x >> 5;
    int lane = threadIdx.x & 31;
    int m0   = blockIdx.x * 16;
    int r    = lane & 15;
    int hi   = lane >> 4;
    const u16* arow = A + (size_t)(m0 + r) * FDIM;

    v8f acc0 = {}; v8f acc1 = {};
    int nt0 = wave * 2, nt1 = wave * 2 + 1;
#pragma unroll
    for (int kt = 0; kt < 8; ++kt) {
        // A: lane<16 = row m0+r, K chunks {kt*32+0..7, kt*32+16..23};
        //    lane>=16 shifts both chunks by 8 (ISA 16-bit A layout).
        int k0 = kt * 32 + hi * 8;
        Frag a, b0, b1;
        a.u[0] = *(const uint4*)(arow + k0);
        a.u[1] = *(const uint4*)(arow + k0 + 16);
        const u16* bp0 = Bp + (((size_t)(nt0 * 8 + kt) * 32 + lane) * 16);
        const u16* bp1 = Bp + (((size_t)(nt1 * 8 + kt) * 32 + lane) * 16);
        b0.u[0] = *(const uint4*)(bp0);
        b0.u[1] = *(const uint4*)(bp0 + 8);
        b1.u[0] = *(const uint4*)(bp1);
        b1.u[1] = *(const uint4*)(bp1 + 8);
        acc0 = __builtin_amdgcn_wmma_f32_16x16x32_bf16(
                   false, a.v, false, b0.v, (short)0, acc0, false, false);
        acc1 = __builtin_amdgcn_wmma_f32_16x16x32_bf16(
                   false, a.v, false, b1.v, (short)0, acc1, false, false);
    }
    // C/D layout: VGPR i -> row m0+i (lanes 0-15), row m0+i+8 (lanes 16-31)
    int col0  = nt0 * 16 + (lane & 15);
    int col1  = nt1 * 16 + (lane & 15);
    int rbase = m0 + hi * 8;
#pragma unroll
    for (int i = 0; i < 8; ++i) {
        C[(size_t)(rbase + i) * FDIM + col0] = acc0[i];
        C[(size_t)(rbase + i) * FDIM + col1] = acc1[i];
    }
}

// --------------------- attention logits: e_src / e_dst ----------------------
// One wave per node; lane l covers 8 contiguous features; heads are 8-lane
// groups (64 features / head) reduced with shfl_xor.
__global__ void attn_logits(const float* __restrict__ h,
                            const float* __restrict__ a_src,
                            const float* __restrict__ a_dst,
                            float* __restrict__ es, float* __restrict__ ed, int N) {
    int wid  = (blockIdx.x * blockDim.x + threadIdx.x) >> 5;
    int lane = threadIdx.x & 31;
    if (wid >= N) return;
    const float4* hp  = (const float4*)(h + (size_t)wid * FDIM + lane * 8);
    const float4* asp = (const float4*)(a_src + lane * 8);
    const float4* adp = (const float4*)(a_dst + lane * 8);
    float4 v0 = hp[0],  v1 = hp[1];
    float ps = dot8(v0, v1, asp[0], asp[1]);
    float pd = dot8(v0, v1, adp[0], adp[1]);
#pragma unroll
    for (int off = 1; off < 8; off <<= 1) {
        ps += __shfl_xor(ps, off, 32);
        pd += __shfl_xor(pd, off, 32);
    }
    if ((lane & 7) == 0) {
        es[wid * HEADS + (lane >> 3)] = ps;
        ed[wid * HEADS + (lane >> 3)] = pd;
    }
}

// --------------------------- edge-wise softmax ------------------------------
__global__ void edge_max(const int* __restrict__ src, const int* __restrict__ dst,
                         int E, int N, const float* __restrict__ es,
                         const float* __restrict__ ed, unsigned* __restrict__ mb) {
    int e = blockIdx.x * blockDim.x + threadIdx.x;
    if (e >= E + N) return;
    int s = e < E ? src[e] : e - E;   // self-loops appended after real edges
    int d = e < E ? dst[e] : e - E;
#pragma unroll
    for (int hh = 0; hh < HEADS; ++hh) {
        float v = es[s * HEADS + hh] + ed[d * HEADS + hh];
        v = v > 0.f ? v : NEG_SLOPE * v;
        atomicMax(&mb[d * HEADS + hh], f2ord(v));
    }
}

__global__ void edge_exp(const int* __restrict__ src, const int* __restrict__ dst,
                         int E, int N, const float* __restrict__ es,
                         const float* __restrict__ ed, const unsigned* __restrict__ mb,
                         float* __restrict__ ee, float* __restrict__ den) {
    int e = blockIdx.x * blockDim.x + threadIdx.x;
    if (e >= E + N) return;
    int s = e < E ? src[e] : e - E;
    int d = e < E ? dst[e] : e - E;
#pragma unroll
    for (int hh = 0; hh < HEADS; ++hh) {
        float v = es[s * HEADS + hh] + ed[d * HEADS + hh];
        v = v > 0.f ? v : NEG_SLOPE * v;
        float ex = expf(v - ord2f(mb[d * HEADS + hh]));
        ee[(size_t)e * HEADS + hh] = ex;
        atomicAdd(&den[d * HEADS + hh], ex);
    }
}

// One wave per edge: lane l moves 8 features of h[src] scaled by alpha into
// agg[dst] via fp32 atomic adds (dominant traffic of the whole model).
__global__ void edge_agg(const int* __restrict__ src, const int* __restrict__ dst,
                         int E, int N, const float* __restrict__ ee,
                         const float* __restrict__ den,
                         const float* __restrict__ h, float* __restrict__ agg) {
    int wid  = (blockIdx.x * blockDim.x + threadIdx.x) >> 5;
    int lane = threadIdx.x & 31;
    if (wid >= E + N) return;
    int s = wid < E ? src[wid] : wid - E;
    int d = wid < E ? dst[wid] : wid - E;
    int head = lane >> 3;
    float alpha = ee[(size_t)wid * HEADS + head] / den[d * HEADS + head];
    const float4* hp = (const float4*)(h + (size_t)s * FDIM + lane * 8);
    float4 v0 = hp[0], v1 = hp[1];
    float* ap = agg + (size_t)d * FDIM + lane * 8;
    atomicAdd(ap + 0, alpha * v0.x);
    atomicAdd(ap + 1, alpha * v0.y);
    atomicAdd(ap + 2, alpha * v0.z);
    atomicAdd(ap + 3, alpha * v0.w);
    atomicAdd(ap + 4, alpha * v1.x);
    atomicAdd(ap + 5, alpha * v1.y);
    atomicAdd(ap + 6, alpha * v1.z);
    atomicAdd(ap + 7, alpha * v1.w);
}

// ------------------------------- BatchNorm ----------------------------------
__global__ void bn_stats(const float* __restrict__ agg, float* __restrict__ sums,
                         float* __restrict__ sums2, int N) {
    int ch = threadIdx.x;                 // 256 channels
    int r0 = blockIdx.x * 128;
    int r1 = min(r0 + 128, N);
    float s = 0.f, s2 = 0.f;
    for (int r = r0; r < r1; ++r) {
        float v = agg[(size_t)r * FDIM + ch];
        s += v; s2 += v * v;
    }
    atomicAdd(&sums[ch], s);
    atomicAdd(&sums2[ch], s2);
}

// y = relu(gamma*(x-mu)*rsqrt(var+eps)+beta), written as bf16 for the next
// GEMMs. (Bias b1 cancels inside the normalization: mu shifts identically.)
__global__ void bn_apply_relu_bf16(const float* __restrict__ agg,
                                   const float* __restrict__ sums,
                                   const float* __restrict__ sums2,
                                   const float* __restrict__ gamma,
                                   const float* __restrict__ beta,
                                   u16* __restrict__ out, int N) {
    int idx = blockIdx.x * 256 + threadIdx.x;       // N*256 threads
    int ch  = idx & (FDIM - 1);
    float inv = 1.f / (float)N;
    float mu  = sums[ch] * inv;
    float var = sums2[ch] * inv - mu * mu;
    float y = (agg[idx] - mu) * rsqrtf(var + BN_EPS) * gamma[ch] + beta[ch];
    out[idx] = f2bf(fmaxf(y, 0.f));
}

// concat=False: mean over the 4 heads + bias -> [N, 64]
__global__ void head_mean(const float* __restrict__ agg, const float* __restrict__ b,
                          float* __restrict__ out, int N) {
    int idx = blockIdx.x * blockDim.x + threadIdx.x;
    if (idx >= N * 64) return;
    int node = idx >> 6, c = idx & 63;
    const float* p = agg + (size_t)node * FDIM + c;
    out[idx] = 0.25f * (p[0] + p[64] + p[128] + p[192]) + b[c];
}

// ------------------------------ orchestration -------------------------------
extern "C" void kernel_launch(void* const* d_in, const int* in_sizes, int n_in,
                              void* d_out, int out_size, void* d_ws, size_t ws_size,
                              hipStream_t stream) {
    const float* x      = (const float*)d_in[0];
    const int*   ei     = (const int*)d_in[1];
    const float* W1     = (const float*)d_in[2];
    const float* a_src1 = (const float*)d_in[3];
    const float* a_dst1 = (const float*)d_in[4];
    /* d_in[5] = b1 : cancels inside BatchNorm */
    const float* gamma1 = (const float*)d_in[6];
    const float* beta1  = (const float*)d_in[7];
    const float* Wm     = (const float*)d_in[8];
    const float* am_src = (const float*)d_in[9];
    const float* am_dst = (const float*)d_in[10];
    const float* bm     = (const float*)d_in[11];
    const float* Wsm    = (const float*)d_in[12];
    const float* as_src = (const float*)d_in[13];
    const float* as_dst = (const float*)d_in[14];
    const float* bs     = (const float*)d_in[15];
    (void)n_in; (void)out_size; (void)ws_size;

    const int N    = in_sizes[0] / FDIM;   // 50000 (divisible by 16)
    const int E    = in_sizes[1] / 2;      // 800000
    const int Etot = E + N;
    const int* src = ei;
    const int* dst = ei + E;

    // workspace carve (~146 MB total)
    char* p = (char*)d_ws;
    auto carve = [&](size_t bytes) -> char* {
        char* r = p; p += (bytes + 255) & ~(size_t)255; return r;
    };
    u16*   xbf   = (u16*)carve((size_t)N * FDIM * 2);   // bf16 layer input
    u16*   wp1   = (u16*)carve((size_t)FDIM * FDIM * 2);
    u16*   wpm   = (u16*)carve((size_t)FDIM * FDIM * 2);
    u16*   wps   = (u16*)carve((size_t)FDIM * FDIM * 2);
    float* hbuf  = (float*)carve((size_t)N * FDIM * 4);
    float* agg   = (float*)carve((size_t)N * FDIM * 4);
    float* es    = (float*)carve((size_t)N * HEADS * 4);
    float* ed    = (float*)carve((size_t)N * HEADS * 4);
    unsigned* mb = (unsigned*)carve((size_t)N * HEADS * 4);
    float* den   = (float*)carve((size_t)N * HEADS * 4);
    float* eew   = (float*)carve((size_t)Etot * HEADS * 4);
    float* sums  = (float*)carve(FDIM * 4);
    float* sums2 = (float*)carve(FDIM * 4);

    const int blkE  = (Etot + 255) / 256;   // thread-per-edge kernels
    const int blkEW = (Etot + 7) / 8;       // wave-per-edge kernels
    const int blkNW = (N + 7) / 8;          // wave-per-node kernels

    cvt_bf16<<<N, 256, 0, stream>>>(x, xbf, N * FDIM);
    pack_w<<<256, 256, 0, stream>>>(W1,  wp1);
    pack_w<<<256, 256, 0, stream>>>(Wm,  wpm);
    pack_w<<<256, 256, 0, stream>>>(Wsm, wps);

    auto gat = [&](const u16* Wp, const float* asv, const float* adv) {
        hipMemsetAsync(agg, 0, (size_t)N * FDIM * 4, stream);
        hipMemsetAsync(mb,  0, (size_t)N * HEADS * 4, stream);
        hipMemsetAsync(den, 0, (size_t)N * HEADS * 4, stream);
        gemm_bf16_wmma<<<N / 16, 256, 0, stream>>>(xbf, Wp, hbuf);
        attn_logits<<<blkNW, 256, 0, stream>>>(hbuf, asv, adv, es, ed, N);
        edge_max<<<blkE, 256, 0, stream>>>(src, dst, E, N, es, ed, mb);
        edge_exp<<<blkE, 256, 0, stream>>>(src, dst, E, N, es, ed, mb, eew, den);
        edge_agg<<<blkEW, 256, 0, stream>>>(src, dst, E, N, eew, den, hbuf, agg);
    };

    // ---- layer 1: GAT(concat) -> BatchNorm -> ReLU (result back into xbf)
    gat(wp1, a_src1, a_dst1);
    hipMemsetAsync(sums,  0, FDIM * 4, stream);
    hipMemsetAsync(sums2, 0, FDIM * 4, stream);
    bn_stats<<<(N + 127) / 128, 256, 0, stream>>>(agg, sums, sums2, N);
    bn_apply_relu_bf16<<<N, 256, 0, stream>>>(agg, sums, sums2, gamma1, beta1, xbf, N);

    float* out = (float*)d_out;
    // ---- conv_mean (concat=False)
    gat(wpm, am_src, am_dst);
    head_mean<<<(N * 64 + 255) / 256, 256, 0, stream>>>(agg, bm, out, N);
    // ---- conv_log_std (concat=False)
    gat(wps, as_src, as_dst);
    head_mean<<<(N * 64 + 255) / 256, 256, 0, stream>>>(agg, bs, out + (size_t)N * 64, N);
}
